// NTM_16870631539244
// MI455X (gfx1250) — matrix operand order repaired and checked by reading, not compiled, not run on hardware
//
#include <hip/hip_runtime.h>
#include <hip/hip_bf16.h>
#include <math.h>

// ---------------------------------------------------------------------------
// NTM reference mapped to MI455X (gfx1250):
//   K0  prep:       f32->f16 conversion of X, Wc; pack 4 head weight mats
//                   into Whh[1024][80] (f16) and biases into bias80[80].
//   K1  gemm_hidden: H = relu(X @ Wc + bc)  via v_wmma_f32_16x16x32_f16,
//                   output stored directly as f16 (feeds K2 only).
//   K2  gemm_keys:  pre = H @ Whh + bias80; per-column activation
//                   (tanh / sigmoid / tanh / tanh) -> keys[1024][80] f32.
//   K3  scan:       single 1024-thread workgroup, memory[8192][20] f32 in
//                   workspace (L2-resident). Thread t owns rows t+1024*i,
//                   so only LDS reductions need barriers.
// ---------------------------------------------------------------------------

#define SEQ_LEN    1024
#define INPUT_DIM  512
#define HIDDEN_DIM 1024
#define MEM_SIZE   8192
#define MEM_DIM    20
#define BETA       2.0f
#define EPS        1e-6f

typedef __attribute__((ext_vector_type(16))) _Float16 v16h;
typedef __attribute__((ext_vector_type(8)))  _Float16 v8h;
typedef __attribute__((ext_vector_type(8)))  float    v8f;

union Frag16 { v16h v; struct { v8h lo, hi; } h; };

// ------------------------------- K0: prep ----------------------------------
__global__ void __launch_bounds__(256) ntm_prep(
    const float* __restrict__ x,  const float* __restrict__ Wc,
    const float* __restrict__ Wrk, const float* __restrict__ We,
    const float* __restrict__ Wa,  const float* __restrict__ Wwk,
    const float* __restrict__ brk, const float* __restrict__ be,
    const float* __restrict__ ba,  const float* __restrict__ bwk,
    _Float16* __restrict__ Xh, _Float16* __restrict__ Wch,
    _Float16* __restrict__ Whh, float* __restrict__ bias80) {
  int idx = blockIdx.x * blockDim.x + threadIdx.x;
  if (idx < SEQ_LEN * INPUT_DIM)  Xh[idx]  = (_Float16)x[idx];
  if (idx < INPUT_DIM * HIDDEN_DIM) Wch[idx] = (_Float16)Wc[idx];
  if (idx < HIDDEN_DIM * 80) {
    int h = idx / 80, c = idx % 80, g = c / MEM_DIM, d = c % MEM_DIM;
    const float* W = (g == 0) ? Wrk : (g == 1) ? We : (g == 2) ? Wa : Wwk;
    Whh[idx] = (_Float16)W[h * MEM_DIM + d];
  }
  if (idx < 80) {
    int g = idx / MEM_DIM, d = idx % MEM_DIM;
    const float* B = (g == 0) ? brk : (g == 1) ? be : (g == 2) ? ba : bwk;
    bias80[idx] = B[d];
  }
}

// --------------------------- K1: H = relu(XWc+b) ---------------------------
// One wave per 16x16 output tile; K=512 in 16 wmma steps.
__global__ void __launch_bounds__(128) ntm_gemm_hidden(
    const _Float16* __restrict__ Xh, const _Float16* __restrict__ Wch,
    const float* __restrict__ bc, _Float16* __restrict__ Hh) {
  const int lane = threadIdx.x & 31;
  const int wid  = threadIdx.x >> 5;
  const int tile = blockIdx.x * 4 + wid;     // 64x64 = 4096 tiles
  const int tm = tile >> 6;
  const int tn = tile & 63;
  const int half = lane >> 4;
  const int mrow = tm * 16 + (lane & 15);

  v8f c = {};
  for (int kb = 0; kb < INPUT_DIM; kb += 32) {
    Frag16 a, b;
    // A 16x32 f16: lanes 0-15 -> K = kb+[0..7], kb+[16..23]; lanes 16-31: +8
    const _Float16* ap = Xh + mrow * INPUT_DIM + kb + half * 8;
    a.h.lo = *(const v8h*)(ap);
    a.h.hi = *(const v8h*)(ap + 16);
    // B 32x16 f16: lane -> column n = lane%16, K order same as A
    const _Float16* bp = Wch + (kb + half * 8) * HIDDEN_DIM + tn * 16 + (lane & 15);
    for (int s = 0; s < 8; ++s) {
      b.h.lo[s] = bp[s * HIDDEN_DIM];
      b.h.hi[s] = bp[(16 + s) * HIDDEN_DIM];
    }
    c = __builtin_amdgcn_wmma_f32_16x16x32_f16(false, a.v, false, b.v,
                                               (short)0, c, false, false);
  }
  const int ncol = tn * 16 + (lane & 15);
  const float bias = bc[ncol];
#pragma unroll
  for (int r = 0; r < 8; ++r) {
    int m = tm * 16 + r + half * 8;
    float v = fmaxf(c[r] + bias, 0.f);
    Hh[m * HIDDEN_DIM + ncol] = (_Float16)v;
  }
}

// ----------------------- K2: keys = act(H @ Whh + b) -----------------------
__global__ void __launch_bounds__(128) ntm_gemm_keys(
    const _Float16* __restrict__ Hh, const _Float16* __restrict__ Whh,
    const float* __restrict__ bias80, float* __restrict__ keys) {
  const int lane = threadIdx.x & 31;
  const int wid  = threadIdx.x >> 5;
  const int tile = blockIdx.x * 4 + wid;     // 64 * 5 = 320 tiles
  const int tm = tile / 5;
  const int tn = tile % 5;
  const int half = lane >> 4;
  const int mrow = tm * 16 + (lane & 15);

  v8f c = {};
  for (int kb = 0; kb < HIDDEN_DIM; kb += 32) {
    Frag16 a, b;
    const _Float16* ap = Hh + mrow * HIDDEN_DIM + kb + half * 8;
    a.h.lo = *(const v8h*)(ap);
    a.h.hi = *(const v8h*)(ap + 16);
    const _Float16* bp = Whh + (kb + half * 8) * 80 + tn * 16 + (lane & 15);
    for (int s = 0; s < 8; ++s) {
      b.h.lo[s] = bp[s * 80];
      b.h.hi[s] = bp[(16 + s) * 80];
    }
    c = __builtin_amdgcn_wmma_f32_16x16x32_f16(false, a.v, false, b.v,
                                               (short)0, c, false, false);
  }
  const int col = tn * 16 + (lane & 15);
  const float bias = bias80[col];
  const int g = col / MEM_DIM;               // 0:rkey 1:erase 2:add 3:wkey
#pragma unroll
  for (int r = 0; r < 8; ++r) {
    int m = tm * 16 + r + half * 8;
    float pre = c[r] + bias;
    float v = (g == 1) ? (1.f / (1.f + expf(-pre))) : tanhf(pre);
    keys[m * 80 + col] = v;
  }
}

// ------------------------------ helpers ------------------------------------
__device__ inline float waveMax(float v) {
#pragma unroll
  for (int off = 16; off; off >>= 1) v = fmaxf(v, __shfl_xor(v, off, 32));
  return v;
}
__device__ inline float waveSum(float v) {
#pragma unroll
  for (int off = 16; off; off >>= 1) v += __shfl_xor(v, off, 32);
  return v;
}

// ------------------------------ K3: scan -----------------------------------
// One workgroup of 1024 threads; thread owns rows {tid + 1024*i, i<8}.
__global__ void __launch_bounds__(1024) ntm_scan(
    const float* __restrict__ keysG, const float* __restrict__ memInit,
    float* __restrict__ Mem, float* __restrict__ out) {
  __shared__ float keyS[80];     // [rkey | erase | add | wkey]
  __shared__ float readS[MEM_DIM];
  __shared__ float accS[MEM_DIM];
  __shared__ float redS[64];
  __shared__ float statS[6];     // 0:maxR 1:maxW 2:sumR 3:sumW 4:B/nkR 5:B/nkW

  const int tid  = threadIdx.x;
  const int lane = tid & 31;
  const int wid  = tid >> 5;

  // copy memory_init -> workspace (inputs must not be mutated)
  {
    const float4* s = (const float4*)memInit;
    float4* d = (float4*)Mem;
    for (int i = tid; i < (MEM_SIZE * MEM_DIM) / 4; i += 1024) d[i] = s[i];
  }
  if (tid < MEM_DIM) accS[tid] = 0.f;
  __syncthreads();

  float lr[8], lw[8];

  for (int t = 0; t < SEQ_LEN; ++t) {
    if (tid < 80) keyS[tid] = keysG[t * 80 + tid];
    if (tid >= 128 && tid < 128 + MEM_DIM) readS[tid - 128] = 0.f;
    if (tid == 256 && t + 1 < SEQ_LEN)
      __builtin_prefetch(keysG + (t + 1) * 80, 0, 0);   // global_prefetch_b8
    __syncthreads();
    if (tid == 0) {
      float s0 = 0.f, s1 = 0.f;
      for (int d = 0; d < MEM_DIM; ++d) {
        s0 += keyS[d] * keyS[d];
        s1 += keyS[60 + d] * keyS[60 + d];
      }
      statS[4] = BETA / fmaxf(sqrtf(s0), EPS);
      statS[5] = BETA / fmaxf(sqrtf(s1), EPS);
    }
    __syncthreads();

    // ---- pass 1: logits for both heads + row norms ----
    const float bONr = statS[4], bONw = statS[5];
    float lmr = -3.4e38f, lmw = -3.4e38f;
#pragma unroll
    for (int i = 0; i < 8; ++i) {
      const float4* row4 = (const float4*)(Mem + (i * 1024 + tid) * MEM_DIM);
      float4 mv[5];
#pragma unroll
      for (int j = 0; j < 5; ++j) mv[j] = row4[j];
      const float* m = (const float*)mv;
      float dr = 0.f, dw = 0.f, s2 = 0.f;
#pragma unroll
      for (int d = 0; d < MEM_DIM; ++d) {
        dr += m[d] * keyS[d];
        dw += m[d] * keyS[60 + d];
        s2 += m[d] * m[d];
      }
      float inm = 1.f / fmaxf(sqrtf(s2), EPS);
      lr[i] = dr * bONr * inm;
      lw[i] = dw * bONw * inm;
      lmr = fmaxf(lmr, lr[i]);
      lmw = fmaxf(lmw, lw[i]);
    }
    // workgroup max
    lmr = waveMax(lmr); lmw = waveMax(lmw);
    if (lane == 0) { redS[wid] = lmr; redS[32 + wid] = lmw; }
    __syncthreads();
    if (tid < 32)            { float v = waveMax(redS[tid]); if (tid == 0)  statS[0] = v; }
    else if (tid < 64)       { float v = waveMax(redS[tid]); if (tid == 32) statS[1] = v; }
    __syncthreads();

    // ---- exp + workgroup sum ----
    const float Mr = statS[0], Mw = statS[1];
    float sr = 0.f, sw = 0.f;
#pragma unroll
    for (int i = 0; i < 8; ++i) {
      lr[i] = expf(lr[i] - Mr); sr += lr[i];
      lw[i] = expf(lw[i] - Mw); sw += lw[i];
    }
    sr = waveSum(sr); sw = waveSum(sw);
    if (lane == 0) { redS[wid] = sr; redS[32 + wid] = sw; }
    __syncthreads();
    if (tid < 32)            { float v = waveSum(redS[tid]); if (tid == 0)  statS[2] = v; }
    else if (tid < 64)       { float v = waveSum(redS[tid]); if (tid == 32) statS[3] = v; }
    __syncthreads();

    // ---- pass 2: read accumulation + memory update (rows owned per-thread)
    const float invSr = 1.f / statS[2];
    const float invSw = 1.f / statS[3];
    float rvec[MEM_DIM];
#pragma unroll
    for (int d = 0; d < MEM_DIM; ++d) rvec[d] = 0.f;
#pragma unroll
    for (int i = 0; i < 8; ++i) {
      float4* row4 = (float4*)(Mem + (i * 1024 + tid) * MEM_DIM);
      float4 mv[5];
#pragma unroll
      for (int j = 0; j < 5; ++j) mv[j] = row4[j];
      float* m = (float*)mv;
      const float rwv = lr[i] * invSr;
      const float wwv = lw[i] * invSw;
#pragma unroll
      for (int d = 0; d < MEM_DIM; ++d) {
        rvec[d] += rwv * m[d];
        m[d] = m[d] * (1.f - wwv * keyS[20 + d]) + wwv * keyS[40 + d];
      }
#pragma unroll
      for (int j = 0; j < 5; ++j) row4[j] = mv[j];
    }
    // reduce 20-dim read vector: wave shuffle then ds_add_f32
#pragma unroll
    for (int d = 0; d < MEM_DIM; ++d) {
      float v = waveSum(rvec[d]);
      if (lane == 0) atomicAdd(&readS[d], v);
    }
    __syncthreads();
    if (tid < MEM_DIM) accS[tid] += readS[tid];
    __syncthreads();
  }

  if (tid < MEM_DIM) out[tid] = accS[tid];
}

// ---------------------------------------------------------------------------
extern "C" void kernel_launch(void* const* d_in, const int* in_sizes, int n_in,
                              void* d_out, int out_size, void* d_ws, size_t ws_size,
                              hipStream_t stream) {
  const float* x       = (const float*)d_in[0];
  const float* Wc      = (const float*)d_in[1];
  const float* bc      = (const float*)d_in[2];
  const float* Wrk     = (const float*)d_in[3];
  const float* brk     = (const float*)d_in[4];
  const float* We      = (const float*)d_in[5];
  const float* be      = (const float*)d_in[6];
  const float* Wa      = (const float*)d_in[7];
  const float* ba      = (const float*)d_in[8];
  const float* Wwk     = (const float*)d_in[9];
  const float* bwk     = (const float*)d_in[10];
  const float* memInit = (const float*)d_in[11];
  float* out = (float*)d_out;

  char* p = (char*)d_ws;
  _Float16* Xh   = (_Float16*)p; p += (size_t)SEQ_LEN * INPUT_DIM * 2;    // 1 MB
  _Float16* Wch  = (_Float16*)p; p += (size_t)INPUT_DIM * HIDDEN_DIM * 2; // 1 MB
  _Float16* Hh   = (_Float16*)p; p += (size_t)SEQ_LEN * HIDDEN_DIM * 2;   // 2 MB
  _Float16* Whh  = (_Float16*)p; p += (size_t)HIDDEN_DIM * 80 * 2;        // 160 KB
  float* bias80  = (float*)p;    p += 512;
  float* keys    = (float*)p;    p += (size_t)SEQ_LEN * 80 * 4;           // 320 KB
  float* Mem     = (float*)p;    p += (size_t)MEM_SIZE * MEM_DIM * 4;     // 640 KB

  ntm_prep<<<(SEQ_LEN * INPUT_DIM + 255) / 256, 256, 0, stream>>>(
      x, Wc, Wrk, We, Wa, Wwk, brk, be, ba, bwk, Xh, Wch, Whh, bias80);

  // 4096 tiles / 4 waves per block
  ntm_gemm_hidden<<<1024, 128, 0, stream>>>(Xh, Wch, bc, Hh);

  // 320 tiles / 4 waves per block
  ntm_gemm_keys<<<80, 128, 0, stream>>>(Hh, Whh, bias80, keys);

  ntm_scan<<<1, 1024, 0, stream>>>(keys, memInit, Mem, out);

  (void)in_sizes; (void)n_in; (void)out_size; (void)ws_size;
}
